// EMDLossCalculator_5514738008418
// MI455X (gfx1250) — compile-verified, matrix operand order checked by reading, not run on hardware
//
#include <hip/hip_runtime.h>
#include <hip/hip_bf16.h>

// EMD / Sinkhorn loss for MI455X (gfx1250, wave32, WMMA).
// N=16 images, C=640 channels, HW=24*24=576 nodes, 256 pairs.

#define N_IMG  16
#define C_CH   640
#define HW     576
#define NPAIR  256
#define NITER  100
#define REG_INV 10.0f   // 1 / reg, reg = 0.1

typedef __attribute__((ext_vector_type(16))) _Float16 v16h;
typedef __attribute__((ext_vector_type(8)))  float    v8f;

union Frag16 { v16h v; uint4 q[2]; };

__device__ __forceinline__ unsigned short f2bf(float f) {
  union { float f; unsigned u; } c; c.f = f;
  return (unsigned short)(c.u >> 16);
}
__device__ __forceinline__ float bf2f(unsigned short h) {
  union { unsigned u; float f; } c; c.u = ((unsigned)h) << 16;
  return c.f;
}

// ---------------------------------------------------------------------------
// Kernel 1: per-(image,node) channel L2-normalize, write transposed f16
// layout [img][node][c] so GEMM A and B fragments are contiguous in K (=c).
// grid (576, 16, 2) ; block 256. z=0 -> x, z=1 -> y.
// ---------------------------------------------------------------------------
__global__ void emd_normalize_kernel(const float* __restrict__ x,
                                     const float* __restrict__ y,
                                     _Float16* __restrict__ xnT,
                                     _Float16* __restrict__ ynT) {
  const int node = blockIdx.x;
  const int img  = blockIdx.y;
  const float* in = (blockIdx.z == 0) ? x : y;
  _Float16* out   = (blockIdx.z == 0) ? xnT : ynT;
  const int tid = threadIdx.x;

  __shared__ float red[256];
  float ss = 0.f;
  for (int c = tid; c < C_CH; c += 256) {
    float v = in[(size_t)img * C_CH * HW + (size_t)c * HW + node];
    ss += v * v;
  }
  red[tid] = ss;
  __syncthreads();
  for (int s = 128; s > 0; s >>= 1) {
    if (tid < s) red[tid] += red[tid + s];
    __syncthreads();
  }
  const float inv = 1.0f / fmaxf(sqrtf(red[0]), 1e-8f);
  for (int c = tid; c < C_CH; c += 256) {
    float v = in[(size_t)img * C_CH * HW + (size_t)c * HW + node];
    out[((size_t)img * HW + node) * C_CH + c] = (_Float16)(v * inv);
  }
}

// ---------------------------------------------------------------------------
// Kernel 2: per-pair GEMM sim = ynT[a] * xnT[b]^T via v_wmma_f32_16x16x32_f16,
// fused Gibbs kernel K = exp((sim-1)*10), stored bf16.
// grid (81, 256): 9x9 64x64 tiles per pair. block 128 = 4 waves.
// Wave grid 2x2; each wave owns a 32x32 sub-block = 2 A frags x 2 B frags
// -> 4 WMMAs per k-step with 1 fragment load per WMMA.
// ---------------------------------------------------------------------------
__global__ void emd_gibbs_gemm_kernel(const _Float16* __restrict__ xnT,
                                      const _Float16* __restrict__ ynT,
                                      unsigned short* __restrict__ Kmat) {
  const int P    = blockIdx.y;
  const int aImg = P >> 4;          // row side: yn[a]
  const int bImg = P & 15;          // col side: xn[b]
  const int tY   = blockIdx.x / 9;
  const int tX   = blockIdx.x % 9;
  const int p0   = tY * 64;
  const int q0   = tX * 64;

  const int tid  = threadIdx.x;     // 128
  const int lane = tid & 31;
  const int wave = tid >> 5;        // 0..3
  const int tr0  = (wave & 1) * 2;  // row 16-tiles {tr0, tr0+1}
  const int tc0  = (wave >> 1) * 2; // col 16-tiles {tc0, tc0+1}

  __shared__ __align__(16) _Float16 sA[64 * 32];
  __shared__ __align__(16) _Float16 sB[64 * 32];

  // global->LDS staging: each thread copies two 16B segments per matrix.
  // 64 rows x 4 segments (8 halfs) = 256 units; 128 threads -> 2 each.
  const int ldRow0 = tid >> 2;              // 0..31
  const int ldRow1 = ldRow0 + 32;           // 32..63
  const int ldSeg  = tid & 3;               // 0..3
  const size_t gA0 = ((size_t)aImg * HW + p0 + ldRow0) * C_CH + ldSeg * 8;
  const size_t gA1 = ((size_t)aImg * HW + p0 + ldRow1) * C_CH + ldSeg * 8;
  const size_t gB0 = ((size_t)bImg * HW + q0 + ldRow0) * C_CH + ldSeg * 8;
  const size_t gB1 = ((size_t)bImg * HW + q0 + ldRow1) * C_CH + ldSeg * 8;

  v8f c00 = {}, c01 = {}, c10 = {}, c11 = {};

  for (int kk = 0; kk < C_CH; kk += 32) {
    *reinterpret_cast<uint4*>(&sA[ldRow0 * 32 + ldSeg * 8]) =
        *reinterpret_cast<const uint4*>(&ynT[gA0 + kk]);
    *reinterpret_cast<uint4*>(&sA[ldRow1 * 32 + ldSeg * 8]) =
        *reinterpret_cast<const uint4*>(&ynT[gA1 + kk]);
    *reinterpret_cast<uint4*>(&sB[ldRow0 * 32 + ldSeg * 8]) =
        *reinterpret_cast<const uint4*>(&xnT[gB0 + kk]);
    *reinterpret_cast<uint4*>(&sB[ldRow1 * 32 + ldSeg * 8]) =
        *reinterpret_cast<const uint4*>(&xnT[gB1 + kk]);
    __syncthreads();

    // 16-bit A 16x32 fragment layout: lane<16 -> row=lane,  K in {0..7,16..23}
    //                                 lane>=16 -> row=lane-16, K in {8..15,24..31}
    const int r  = lane & 15;
    const int hb = (lane < 16) ? 0 : 8;

    Frag16 fa0, fa1, fb0, fb1;
    const _Float16* pa0 = &sA[(tr0 * 16 + r) * 32 + hb];
    fa0.q[0] = *reinterpret_cast<const uint4*>(pa0);
    fa0.q[1] = *reinterpret_cast<const uint4*>(pa0 + 16);
    const _Float16* pa1 = &sA[((tr0 + 1) * 16 + r) * 32 + hb];
    fa1.q[0] = *reinterpret_cast<const uint4*>(pa1);
    fa1.q[1] = *reinterpret_cast<const uint4*>(pa1 + 16);
    const _Float16* pb0 = &sB[(tc0 * 16 + r) * 32 + hb];
    fb0.q[0] = *reinterpret_cast<const uint4*>(pb0);
    fb0.q[1] = *reinterpret_cast<const uint4*>(pb0 + 16);
    const _Float16* pb1 = &sB[((tc0 + 1) * 16 + r) * 32 + hb];
    fb1.q[0] = *reinterpret_cast<const uint4*>(pb1);
    fb1.q[1] = *reinterpret_cast<const uint4*>(pb1 + 16);

    c00 = __builtin_amdgcn_wmma_f32_16x16x32_f16(false, fa0.v, false, fb0.v,
                                                 (short)0, c00, false, false);
    c01 = __builtin_amdgcn_wmma_f32_16x16x32_f16(false, fa0.v, false, fb1.v,
                                                 (short)0, c01, false, false);
    c10 = __builtin_amdgcn_wmma_f32_16x16x32_f16(false, fa1.v, false, fb0.v,
                                                 (short)0, c10, false, false);
    c11 = __builtin_amdgcn_wmma_f32_16x16x32_f16(false, fa1.v, false, fb1.v,
                                                 (short)0, c11, false, false);
    __syncthreads();
  }

  // C/D layout: VGPR e -> lanes 0-15: (M=e, N=lane); lanes 16-31: (M=8+e, N=lane-16)
  const int rOff = (lane < 16) ? 0 : 8;
  const int colA = q0 + tc0 * 16 + (lane & 15);
  const int colB = colA + 16;
  unsigned short* kp = Kmat + (size_t)P * HW * HW;
  for (int e = 0; e < 8; ++e) {
    const int row0 = p0 + tr0 * 16 + rOff + e;
    const int row1 = row0 + 16;
    kp[(size_t)row0 * HW + colA] = f2bf(__expf((c00[e] - 1.0f) * REG_INV));
    kp[(size_t)row0 * HW + colB] = f2bf(__expf((c01[e] - 1.0f) * REG_INV));
    kp[(size_t)row1 * HW + colA] = f2bf(__expf((c10[e] - 1.0f) * REG_INV));
    kp[(size_t)row1 * HW + colB] = f2bf(__expf((c11[e] - 1.0f) * REG_INV));
  }
}

// ---------------------------------------------------------------------------
// Kernel 3: 100 Sinkhorn iterations + transport cost, one workgroup per pair.
// block 512 = 16 waves. Wave w owns rows {w, w+16, ...} (36 rows).
// K rows read as u32 (2 x bf16 per lane -> 128B per wave per load).
// Forward matvec: coalesced row reads + shuffle reduce.
// Transpose matvec: per-wave LDS partial vectors, then cross-wave combine.
// Cost recovers M = -0.1*log(K) from bf16 K (no extra 170MB for M).
// ---------------------------------------------------------------------------
__global__ void emd_sinkhorn_kernel(const unsigned short* __restrict__ Kmat,
                                    const float* __restrict__ x,
                                    const float* __restrict__ y,
                                    float* __restrict__ costs) {
  const int P = blockIdx.x;
  const int i = P >> 4;
  const int j = P & 15;
  const int tid  = threadIdx.x;   // 512
  const int lane = tid & 31;
  const int wave = tid >> 5;      // 0..15

  __shared__ float sa[HW], sb[HW], su[HW], sv[HW];
  __shared__ float sp[16 * HW];   // per-wave partials for K^T * a

  const unsigned short* K0 = Kmat + (size_t)P * HW * HW;

  // u = relu(x[i, ch j, :]) + 1e-5 ; v = relu(y[j, ch i, :]) + 1e-5 ; b = 1
  for (int q = tid; q < HW; q += 512) {
    su[q] = fmaxf(x[((size_t)i * C_CH + j) * HW + q], 0.f) + 1e-5f;
    sv[q] = fmaxf(y[((size_t)j * C_CH + i) * HW + q], 0.f) + 1e-5f;
    sb[q] = 1.0f;
  }
  __syncthreads();

  for (int it = 0; it <= NITER; ++it) {
    // a = u / (K b)
    for (int row = wave; row < HW; row += 16) {
      const uint* kr = reinterpret_cast<const uint*>(K0 + (size_t)row * HW);
      float s = 0.f;
      for (int t = lane; t < HW / 2; t += 32) {
        const uint w2 = kr[t];
        s += bf2f((unsigned short)(w2 & 0xffffu)) * sb[2 * t] +
             bf2f((unsigned short)(w2 >> 16))     * sb[2 * t + 1];
      }
      for (int off = 16; off; off >>= 1) s += __shfl_down(s, off, 32);
      if (lane == 0) sa[row] = su[row] / s;
    }
    __syncthreads();
    if (it == NITER) break;       // final a-update only (matches reference)

    // b = v / (K^T a)
    float* pw = &sp[wave * HW];
    for (int q = lane; q < HW; q += 32) pw[q] = 0.f;
    for (int row = wave; row < HW; row += 16) {
      const uint* kr = reinterpret_cast<const uint*>(K0 + (size_t)row * HW);
      const float ar = sa[row];   // written by this wave's lane 0
      for (int t = lane; t < HW / 2; t += 32) {
        const uint w2 = kr[t];
        pw[2 * t]     += bf2f((unsigned short)(w2 & 0xffffu)) * ar;
        pw[2 * t + 1] += bf2f((unsigned short)(w2 >> 16))     * ar;
      }
    }
    __syncthreads();
    for (int q = tid; q < HW; q += 512) {
      float s = 0.f;
      for (int w = 0; w < 16; ++w) s += sp[w * HW + q];
      sb[q] = sv[q] / s;
    }
    __syncthreads();
  }

  // cost = sum_i a_i * sum_j K_ij * M_ij * b_j, M_ij = -0.1*log(K_ij)
  float wacc = 0.f;
  for (int row = wave; row < HW; row += 16) {
    const uint* kr = reinterpret_cast<const uint*>(K0 + (size_t)row * HW);
    float s = 0.f;
    for (int t = lane; t < HW / 2; t += 32) {
      const uint w2 = kr[t];
      const float k0 = bf2f((unsigned short)(w2 & 0xffffu));
      const float k1 = bf2f((unsigned short)(w2 >> 16));
      s += k0 * (-0.1f * __logf(fmaxf(k0, 1e-30f))) * sb[2 * t];
      s += k1 * (-0.1f * __logf(fmaxf(k1, 1e-30f))) * sb[2 * t + 1];
    }
    for (int off = 16; off; off >>= 1) s += __shfl_down(s, off, 32);
    if (lane == 0) wacc += sa[row] * s;
  }
  if (lane == 0) sp[wave] = wacc;
  __syncthreads();
  if (tid == 0) {
    float t = 0.f;
    for (int w = 0; w < 16; ++w) t += sp[w];
    costs[P] = t;
  }
}

// ---------------------------------------------------------------------------
// Kernel 4: deterministic final reduction: out = sum(costs)/256
// ---------------------------------------------------------------------------
__global__ void emd_reduce_kernel(const float* __restrict__ costs,
                                  float* __restrict__ out) {
  __shared__ float red[256];
  const int tid = threadIdx.x;
  red[tid] = costs[tid];
  __syncthreads();
  for (int s = 128; s > 0; s >>= 1) {
    if (tid < s) red[tid] += red[tid + s];
    __syncthreads();
  }
  if (tid == 0) out[0] = red[0] / (float)NPAIR;
}

// ---------------------------------------------------------------------------
extern "C" void kernel_launch(void* const* d_in, const int* in_sizes, int n_in,
                              void* d_out, int out_size, void* d_ws, size_t ws_size,
                              hipStream_t stream) {
  (void)in_sizes; (void)n_in; (void)out_size; (void)ws_size;
  const float* x = (const float*)d_in[0];
  const float* y = (const float*)d_in[1];

  char* ws = (char*)d_ws;
  const size_t XN_HALFS = (size_t)N_IMG * HW * C_CH;           // 5,898,240
  _Float16* xnT = (_Float16*)ws;                               // 11.8 MB
  _Float16* ynT = (_Float16*)(ws + XN_HALFS * 2);              // 11.8 MB
  unsigned short* Kmat = (unsigned short*)(ws + 4 * XN_HALFS); // 170 MB bf16
  float* costs = (float*)(ws + 4 * XN_HALFS +
                          (size_t)NPAIR * HW * HW * 2);        // 1 KB

  emd_normalize_kernel<<<dim3(HW, N_IMG, 2), 256, 0, stream>>>(x, y, xnT, ynT);
  emd_gibbs_gemm_kernel<<<dim3(81, NPAIR), 128, 0, stream>>>(xnT, ynT, Kmat);
  emd_sinkhorn_kernel<<<NPAIR, 512, 0, stream>>>(Kmat, x, y, costs);
  emd_reduce_kernel<<<1, 256, 0, stream>>>(costs, (float*)d_out);
}
